// AttentionBlock_74526272520384
// MI455X (gfx1250) — matrix operand order repaired and checked by reading, not compiled, not run on hardware
//
#include <hip/hip_runtime.h>

typedef __bf16 bf16;
typedef __attribute__((ext_vector_type(16))) __bf16 v16bf;
typedef __attribute__((ext_vector_type(8)))  __bf16 v8bf;
typedef __attribute__((ext_vector_type(8)))  float  v8f;

#define WMMA_BF16(A_, B_, C_) \
  __builtin_amdgcn_wmma_f32_16x16x32_bf16(false, (A_), false, (B_), (short)0, (C_), false, false)

__device__ __forceinline__ v16bf cat16(v8bf lo, v8bf hi) {
  v16bf r;
#pragma unroll
  for (int i = 0; i < 8; ++i) { r[i] = lo[i]; r[i + 8] = hi[i]; }
  return r;
}

// A fragment (16x32 bf16, row-major window of 32 K values starting at rowptr).
// halfsel 0 (lanes 0-15): K {0..15}; halfsel 1 (lanes 16-31): K {16..31}.
__device__ __forceinline__ v16bf load_a_frag(const bf16* rowptr, int halfsel) {
  const int o = halfsel ? 16 : 0;
  v8bf lo = *(const v8bf*)(rowptr + o);
  v8bf hi = *(const v8bf*)(rowptr + o + 8);
  return cat16(lo, hi);
}

// B fragment: colptr = &Bs[col*stride + halfsel*16]; 16 contiguous K values.
__device__ __forceinline__ v16bf load_b_frag(const bf16* colptr) {
  v8bf lo = *(const v8bf*)(colptr);
  v8bf hi = *(const v8bf*)(colptr + 8);
  return cat16(lo, hi);
}

// gfx1250 async copy: 16B global -> LDS per lane, tracked by ASYNCcnt.
__device__ __forceinline__ void async_load_b128(unsigned lds_off, unsigned long long gaddr) {
  asm volatile("global_load_async_to_lds_b128 %0, %1, off"
               :: "v"(lds_off), "v"(gaddr) : "memory");
}
__device__ __forceinline__ void wait_async0() {
  asm volatile("s_wait_asynccnt 0" ::: "memory");
}
__device__ __forceinline__ unsigned lds_addr(const void* p) {
  return (unsigned)(unsigned long long)p;   // flat low 32 bits = LDS offset
}

// ---------------------------------------------------------------------------
// fp32 -> bf16 cast (weights)
// ---------------------------------------------------------------------------
__global__ void cast_f32_to_bf16(const float* __restrict__ in, bf16* __restrict__ out, int n) {
  int i = (blockIdx.x * 256 + threadIdx.x) * 4;
  if (i + 3 < n) {
    float4 v = *(const float4*)(in + i);
    out[i] = (bf16)v.x; out[i + 1] = (bf16)v.y;
    out[i + 2] = (bf16)v.z; out[i + 3] = (bf16)v.w;
  } else {
    for (int k = i; k < n && k < i + 4; ++k) out[k] = (bf16)in[k];
  }
}

// ---------------------------------------------------------------------------
// LayerNorm over D=512, output bf16. One block per row.
// ---------------------------------------------------------------------------
__global__ __launch_bounds__(256)
void ln_cast_kernel(const float* __restrict__ x, const float* __restrict__ g,
                    const float* __restrict__ b, bf16* __restrict__ y)
{
  const int D = 512;
  const int row = blockIdx.x;
  const float* xr = x + (size_t)row * D;
  float s0 = 0.f, s1 = 0.f;
  for (int i = threadIdx.x; i < D; i += 256) { float v = xr[i]; s0 += v; s1 += v * v; }
#pragma unroll
  for (int off = 16; off > 0; off >>= 1) {
    s0 += __shfl_xor(s0, off, 32);
    s1 += __shfl_xor(s1, off, 32);
  }
  __shared__ float red0[8], red1[8], stat[2];
  const int wave = threadIdx.x >> 5, lane = threadIdx.x & 31;
  if (lane == 0) { red0[wave] = s0; red1[wave] = s1; }
  __syncthreads();
  if (threadIdx.x == 0) {
    float a = 0.f, c = 0.f;
#pragma unroll
    for (int w = 0; w < 8; ++w) { a += red0[w]; c += red1[w]; }
    float mu = a / D;
    stat[0] = mu;
    stat[1] = rsqrtf(c / D - mu * mu + 1e-5f);
  }
  __syncthreads();
  const float mu = stat[0], rstd = stat[1];
  for (int i = threadIdx.x; i < D; i += 256)
    y[(size_t)row * D + i] = (bf16)((xr[i] - mu) * rstd * g[i] + b[i]);
}

// ---------------------------------------------------------------------------
// bf16 WMMA GEMM: C[M,N] = A[M,K] * B[K,N] (+bias, epilogue)
// block tile 128x128 (8 waves: 4x2), wave tile 32x64 (2x4 wmma), K-step 32.
// A tile staged via global_load_async_to_lds_b128; B tile staged [n][k].
// EPI 0: bias -> bf16 out; EPI 1: bias + resid -> f32 out; EPI 2: bias+swish -> bf16.
// ---------------------------------------------------------------------------
template <int EPI>
__global__ __launch_bounds__(256, 2)
void gemm_bf16_kernel(const bf16* __restrict__ A, const bf16* __restrict__ B,
                      const float* __restrict__ bias, const float* __restrict__ resid,
                      float* __restrict__ outF, bf16* __restrict__ outH,
                      int M, int N, int K)
{
  __shared__ bf16 As[128 * 32];   // [m][k]
  __shared__ bf16 Bs[128 * 32];   // [n][k]  (k-contiguous for B fragments)

  const int tid = threadIdx.x;
  const int lane = tid & 31;
  const int wave = tid >> 5;
  const int wm = wave >> 1, wn = wave & 1;
  const int l16 = lane & 15, halfsel = lane >> 4;
  const int bm0 = blockIdx.x * 128;
  const int bn0 = blockIdx.y * 128;

  v8f acc[2][4] = {};

  const int arow = tid >> 1, apart = tid & 1;          // A: 2 x b128 per thread
  const int bk = tid >> 3, bn = (tid & 7) * 16;        // B: 16 n per thread

  const unsigned la = lds_addr(&As[arow * 32 + apart * 16]);

  for (int k0 = 0; k0 < K; k0 += 32) {
    const bf16* ga = A + (size_t)(bm0 + arow) * K + k0 + apart * 16;
    async_load_b128(la,      (unsigned long long)ga);
    async_load_b128(la + 16, (unsigned long long)(ga + 8));

    const bf16* gb = B + (size_t)(k0 + bk) * N + bn0 + bn;
    v8bf b0 = *(const v8bf*)gb;
    v8bf b1 = *(const v8bf*)(gb + 8);
#pragma unroll
    for (int e = 0; e < 8; ++e) {
      Bs[(bn + e) * 32 + bk]     = b0[e];
      Bs[(bn + 8 + e) * 32 + bk] = b1[e];
    }
    if (k0 + 32 < K) __builtin_prefetch(gb + (size_t)32 * N, 0, 1);

    wait_async0();
    __syncthreads();
#pragma unroll
    for (int i = 0; i < 2; ++i) {
      v16bf af = load_a_frag(&As[(wm * 32 + i * 16 + l16) * 32], halfsel);
#pragma unroll
      for (int j = 0; j < 4; ++j) {
        v16bf bfr = load_b_frag(&Bs[(wn * 64 + j * 16 + l16) * 32 + halfsel * 16]);
        acc[i][j] = WMMA_BF16(af, bfr, acc[i][j]);
      }
    }
    __syncthreads();
  }

#pragma unroll
  for (int i = 0; i < 2; ++i)
#pragma unroll
    for (int j = 0; j < 4; ++j) {
      const int col = bn0 + wn * 64 + j * 16 + l16;
      const float bv = bias[col];
      const int m0 = bm0 + wm * 32 + i * 16 + halfsel * 8;
#pragma unroll
      for (int r = 0; r < 8; ++r) {
        const size_t idx = (size_t)(m0 + r) * N + col;
        float v = acc[i][j][r] + bv;
        if (EPI == 0) {
          outH[idx] = (bf16)v;
        } else if (EPI == 1) {
          outF[idx] = v + resid[idx];
        } else {
          outH[idx] = (bf16)(v / (1.f + __expf(-v)));   // Swish
        }
      }
    }
}

// ---------------------------------------------------------------------------
// Flash attention: one block per (qblock of 32 rows, head, batch).
// 8 waves = 2 q-subtiles (16 rows) x 4 d_v slices (128 cols).
// Q/K tiles staged with async-to-LDS (row-major is fragment-ready); V transposed.
// ---------------------------------------------------------------------------
__global__ __launch_bounds__(256, 1)
void attn_kernel(const bf16* __restrict__ qkv, bf16* __restrict__ aout)
{
  constexpr int Ntok = 2048, QKVS = 5120, DH = 64, DV = 512, HOUT = 4096;
  __shared__ bf16 Qs[32 * 64];
  __shared__ bf16 Ks[32 * 64];
  __shared__ bf16 Vts[512 * 32];      // [col][key] transposed V chunk
  __shared__ bf16 Ps[2][16 * 32];
  __shared__ float mrow[32], lrow[32], crow[32];

  const int tid = threadIdx.x;
  const int lane = tid & 31;
  const int wave = tid >> 5;
  const int qsub = wave >> 2;
  const int dsl  = wave & 3;
  const int l16 = lane & 15, halfsel = lane >> 4;
  const int qb = blockIdx.x;
  const int h  = blockIdx.y;
  const int bz = blockIdx.z;
  const size_t tok0q = (size_t)bz * Ntok + qb * 32;

  const int str = tid >> 3, stc = (tid & 7) * 8;       // 32x64 tile staging
  async_load_b128(lds_addr(&Qs[str * 64 + stc]),
                  (unsigned long long)(qkv + (tok0q + str) * QKVS + h * DH + stc));
  if (tid < 32) { mrow[tid] = -1e30f; lrow[tid] = 0.f; }

  v8f o[8] = {};

  for (int kb = 0; kb < Ntok; kb += 32) {
    __syncthreads();                       // previous consumers done
    const size_t tok0k = (size_t)bz * Ntok + kb;
    async_load_b128(lds_addr(&Ks[str * 64 + stc]),
                    (unsigned long long)(qkv + (tok0k + str) * QKVS + 512 + h * DH + stc));
    {
      int key = tid >> 3, cb = (tid & 7) * 64;
      const bf16* gp = qkv + (tok0k + key) * QKVS + 1024 + h * DV + cb;
#pragma unroll
      for (int u = 0; u < 8; ++u) {
        v8bf vv = *(const v8bf*)(gp + u * 8);
#pragma unroll
        for (int e = 0; e < 8; ++e) Vts[(cb + u * 8 + e) * 32 + key] = vv[e];
      }
    }
    wait_async0();
    __syncthreads();

    if (dsl == 0) {                        // whole wave uniform -> EXEC all-1s for WMMA
      v8f s0 = {}, s1 = {};
#pragma unroll
      for (int ks = 0; ks < 64; ks += 32) {
        v16bf aq  = load_a_frag(&Qs[(qsub * 16 + l16) * 64 + ks], halfsel);
        v16bf bk0 = load_b_frag(&Ks[l16 * 64 + ks + halfsel * 16]);
        v16bf bk1 = load_b_frag(&Ks[(16 + l16) * 64 + ks + halfsel * 16]);
        s0 = WMMA_BF16(aq, bk0, s0);
        s1 = WMMA_BF16(aq, bk1, s1);
      }
#pragma unroll
      for (int r = 0; r < 8; ++r) {
        const int row = qsub * 16 + halfsel * 8 + r;
        float a  = s0[r] * 0.125f;         // 1/sqrt(64)
        float bq = s1[r] * 0.125f;
        float rm = fmaxf(a, bq);
#pragma unroll
        for (int off = 1; off < 16; off <<= 1) rm = fmaxf(rm, __shfl_xor(rm, off, 32));
        const float mold = mrow[row];
        const float mnew = fmaxf(mold, rm);
        const float p0 = __expf(a - mnew);
        const float p1 = __expf(bq - mnew);
        float ps = p0 + p1;
#pragma unroll
        for (int off = 1; off < 16; off <<= 1) ps += __shfl_xor(ps, off, 32);
        const float corr = __expf(mold - mnew);
        if (l16 == 0) {
          mrow[row] = mnew;
          lrow[row] = lrow[row] * corr + ps;
          crow[row] = corr;
        }
        Ps[qsub][(halfsel * 8 + r) * 32 + l16]      = (bf16)p0;
        Ps[qsub][(halfsel * 8 + r) * 32 + 16 + l16] = (bf16)p1;
      }
    }
    __syncthreads();

    float cr[8];
#pragma unroll
    for (int r = 0; r < 8; ++r) cr[r] = crow[qsub * 16 + halfsel * 8 + r];
#pragma unroll
    for (int ct = 0; ct < 8; ++ct)
#pragma unroll
      for (int r = 0; r < 8; ++r) o[ct][r] *= cr[r];

    v16bf ap = load_a_frag(&Ps[qsub][l16 * 32], halfsel);
#pragma unroll
    for (int ct = 0; ct < 8; ++ct) {
      v16bf bv = load_b_frag(&Vts[(dsl * 128 + ct * 16 + l16) * 32 + halfsel * 16]);
      o[ct] = WMMA_BF16(ap, bv, o[ct]);
    }
  }

  float li[8];
#pragma unroll
  for (int r = 0; r < 8; ++r) li[r] = 1.f / lrow[qsub * 16 + halfsel * 8 + r];
  const size_t rowg = tok0q + qsub * 16 + halfsel * 8;
#pragma unroll
  for (int ct = 0; ct < 8; ++ct)
#pragma unroll
    for (int r = 0; r < 8; ++r)
      aout[(rowg + r) * HOUT + h * DV + dsl * 128 + ct * 16 + l16] = (bf16)(o[ct][r] * li[r]);
}

// ---------------------------------------------------------------------------
extern "C" void kernel_launch(void* const* d_in, const int* in_sizes, int n_in,
                              void* d_out, int out_size, void* d_ws, size_t ws_size,
                              hipStream_t stream)
{
  (void)in_sizes; (void)n_in; (void)out_size; (void)ws_size;
  const float* x     = (const float*)d_in[0];
  const float* ln1_g = (const float*)d_in[1];
  const float* ln1_b = (const float*)d_in[2];
  const float* Wqkv  = (const float*)d_in[3];
  const float* bqkv  = (const float*)d_in[4];
  const float* Wm    = (const float*)d_in[5];
  const float* bm    = (const float*)d_in[6];
  const float* ln2_g = (const float*)d_in[7];
  const float* ln2_b = (const float*)d_in[8];
  const float* W1    = (const float*)d_in[9];
  const float* b1    = (const float*)d_in[10];
  const float* W2    = (const float*)d_in[11];
  const float* b2    = (const float*)d_in[12];
  float* outp = (float*)d_out;

  const int Mtok = 4096;                       // B*N rows
  char* base = (char*)d_ws;
  size_t off = 0;
  auto alloc = [&](size_t elems, size_t esz) -> void* {
    void* p = base + off;
    off += (elems * esz + 255) & ~(size_t)255;
    return p;
  };
  bf16* wqkv_h = (bf16*)alloc((size_t)512 * 5120, 2);
  bf16* wm_h   = (bf16*)alloc((size_t)4096 * 512, 2);
  bf16* w1_h   = (bf16*)alloc((size_t)512 * 2048, 2);
  bf16* w2_h   = (bf16*)alloc((size_t)2048 * 512, 2);
  bf16* xln    = (bf16*)alloc((size_t)Mtok * 512, 2);
  bf16* qkv_h  = (bf16*)alloc((size_t)Mtok * 5120, 2);
  bf16* attn_h = (bf16*)alloc((size_t)Mtok * 4096, 2);
  float* x2    = (float*)alloc((size_t)Mtok * 512, 4);
  bf16* x2ln   = (bf16*)alloc((size_t)Mtok * 512, 2);
  bf16* h_h    = (bf16*)alloc((size_t)Mtok * 2048, 2);

  auto cast = [&](const float* src, bf16* dst, int n) {
    cast_f32_to_bf16<<<(n + 1023) / 1024, 256, 0, stream>>>(src, dst, n);
  };
  cast(Wqkv, wqkv_h, 512 * 5120);
  cast(Wm,   wm_h,   4096 * 512);
  cast(W1,   w1_h,   512 * 2048);
  cast(W2,   w2_h,   2048 * 512);

  ln_cast_kernel<<<Mtok, 256, 0, stream>>>(x, ln1_g, ln1_b, xln);

  gemm_bf16_kernel<0><<<dim3(Mtok / 128, 5120 / 128), 256, 0, stream>>>(
      xln, wqkv_h, bqkv, nullptr, nullptr, qkv_h, Mtok, 5120, 512);

  attn_kernel<<<dim3(64, 8, 2), 256, 0, stream>>>(qkv_h, attn_h);

  gemm_bf16_kernel<1><<<dim3(Mtok / 128, 512 / 128), 256, 0, stream>>>(
      attn_h, wm_h, bm, x, x2, nullptr, Mtok, 512, 4096);

  ln_cast_kernel<<<Mtok, 256, 0, stream>>>(x2, ln2_g, ln2_b, x2ln);

  gemm_bf16_kernel<2><<<dim3(Mtok / 128, 2048 / 128), 256, 0, stream>>>(
      x2ln, w1_h, b1, nullptr, nullptr, h_h, Mtok, 2048, 512);

  gemm_bf16_kernel<1><<<dim3(Mtok / 128, 512 / 128), 256, 0, stream>>>(
      h_h, w2_h, b2, x2, outp, nullptr, Mtok, 512, 2048);
}
